// MultiHeadSpatialSelfAttention_64922725646486
// MI455X (gfx1250) — matrix operand order repaired and checked by reading, not compiled, not run on hardware
//
#include <hip/hip_runtime.h>
#include <hip/hip_bf16.h>

typedef __attribute__((ext_vector_type(16))) _Float16 v16h;
typedef __attribute__((ext_vector_type(8)))  _Float16 v8h;
typedef __attribute__((ext_vector_type(8)))  float    v8f;
typedef __attribute__((ext_vector_type(4)))  float    v4f;

#define N_SP   4096
#define C_IN   256
#define HS_DIM 64
#define PR     4100   // padded LDS row stride (floats): 4100%64=4 -> de-conflicted banks, 16B aligned

static __device__ inline v8f wmma16(v16h a, v16h b, v8f c) {
  return __builtin_amdgcn_wmma_f32_16x16x32_f16(false, a, false, b, (short)0, c, false, false);
}

// A fragment (16x32 f16, row-major source). Contiguous layout: lane-lo holds K in
// [0,16), lane-hi holds [16,32) -> two 16B loads per lane.
static __device__ inline v16h mk_afrag(const _Float16* p, int hi) {
  v8h lo = *(const v8h*)(p + hi * 16);
  v8h hh = *(const v8h*)(p + 8 + hi * 16);
  v16h r;
#pragma unroll
  for (int i = 0; i < 8; ++i) { r[i] = lo[i]; r[8 + i] = hh[i]; }
  return r;
}

// C = A(16x256) * B(256x16), A rows / B cols precomputed per lane, ld=256.
static __device__ inline v8f gemm_k256(const _Float16* Arow, const _Float16* Bcol, int hi) {
  v8f acc = {};
#pragma unroll
  for (int kk = 0; kk < 8; ++kk) {
    int k0 = kk * 32;
    v16h a = mk_afrag(Arow + k0, hi);
    v16h b = *(const v16h*)(Bcol + k0 + hi * 16);
    acc = wmma16(a, b, acc);
  }
  return acc;
}

// ---- Kernel 0: transpose+convert x [256][4096] f32 -> xt [4096][256] f16 ----
// x is read exactly once -> non-temporal loads keep it out of L2.
__global__ __launch_bounds__(256) void k_transpose_x(const float* __restrict__ x,
                                                     _Float16* __restrict__ xt) {
  __shared__ float t[32][33];
  int n0 = blockIdx.x * 32, c0 = blockIdx.y * 32;
  int tx = threadIdx.x & 31, ty = threadIdx.x >> 5;
#pragma unroll
  for (int i = 0; i < 4; ++i)
    t[ty + 8 * i][tx] =
        __builtin_nontemporal_load(x + (size_t)(c0 + ty + 8 * i) * N_SP + n0 + tx);
  __syncthreads();
#pragma unroll
  for (int i = 0; i < 4; ++i) {
    int r = ty + 8 * i;
    xt[(size_t)(n0 + r) * C_IN + c0 + tx] = (_Float16)t[tx][r];
  }
}

// ---- Kernel 0b: fold Wf*log2e/TEMP into Q weights, convert weights, fold BN ----
__global__ void k_prep_w(const float* Wq, const float* bq, const float* Wk, const float* Wv,
                         const float* Wf, const float* Wo, const float* bo, const float* gamma,
                         const float* beta, const float* mean, const float* var,
                         _Float16* Wq_s, float* bq_s, _Float16* Wkh, _Float16* Wvh,
                         _Float16* Woh, float* scaleBN, float* shiftBN) {
  int gid = blockIdx.x * blockDim.x + threadIdx.x;
  int T = gridDim.x * blockDim.x;
  const float QS = 1.4426950408889634f / 8.0f;   // log2(e) / sqrt(HS)
  for (int i = gid; i < 4 * 64 * 256; i += T) {
    int o = i >> 14, hs = (i >> 8) & 63, c = i & 255;
    Wq_s[i] = (_Float16)(Wq[hs * 256 + c] * Wf[o * 4 + (hs >> 4)] * QS);
  }
  for (int i = gid; i < 256; i += T) {
    int o = i >> 6, hs = i & 63;
    bq_s[i] = bq[hs] * Wf[o * 4 + (hs >> 4)] * QS;
  }
  for (int i = gid; i < 64 * 256; i += T) { Wkh[i] = (_Float16)Wk[i]; Wvh[i] = (_Float16)Wv[i]; }
  for (int i = gid; i < 256 * 64; i += T) { Woh[i] = (_Float16)Wo[i]; }
  for (int i = gid; i < 256; i += T) {
    float s = gamma[i] * rsqrtf(var[i] + 1e-5f);
    scaleBN[i] = s;
    shiftBN[i] = (bo[i] - mean[i]) * s + beta[i];
  }
}

// ---- Kernel 1: QKV projections (f16 WMMA, K=256). One 16x16 tile per wave. ----
__global__ __launch_bounds__(256) void k_qkv(const _Float16* __restrict__ xt,
                                             const _Float16* __restrict__ Wq_s,
                                             const float* __restrict__ bq_s,
                                             const _Float16* __restrict__ Wkh,
                                             const float* __restrict__ bk,
                                             const _Float16* __restrict__ Wvh,
                                             const float* __restrict__ bv,
                                             _Float16* __restrict__ Qf,
                                             _Float16* __restrict__ Kt,
                                             _Float16* __restrict__ Vh) {
  int wave = blockIdx.x * 8 + (threadIdx.x >> 5);
  int lane = threadIdx.x & 31, col = lane & 15, hi = lane >> 4;
  if (wave < 4096) {                       // Qf[o][n][64]  (M=hs, N=n)
    int o = wave >> 10, t = wave & 1023;
    int n0 = (t >> 2) << 4, hs0 = (t & 3) << 4;
    v8f acc = gemm_k256(Wq_s + (size_t)o * 16384 + (size_t)(hs0 + col) * 256,
                        xt + (size_t)(n0 + col) * 256, hi);
    const float* bias = bq_s + o * 64 + hs0 + 8 * hi;
    v8h ph;
#pragma unroll
    for (int r = 0; r < 8; ++r) ph[r] = (_Float16)(acc[r] + bias[r]);
    *(v8h*)(Qf + (size_t)o * (N_SP * 64) + (size_t)(n0 + col) * 64 + hs0 + 8 * hi) = ph;
  } else if (wave < 5120) {                // Kt[n][64]
    int t = wave - 4096;
    int n0 = (t >> 2) << 4, hs0 = (t & 3) << 4;
    v8f acc = gemm_k256(Wkh + (size_t)(hs0 + col) * 256, xt + (size_t)(n0 + col) * 256, hi);
    const float* bias = bk + hs0 + 8 * hi;
    v8h ph;
#pragma unroll
    for (int r = 0; r < 8; ++r) ph[r] = (_Float16)(acc[r] + bias[r]);
    *(v8h*)(Kt + (size_t)(n0 + col) * 64 + hs0 + 8 * hi) = ph;
  } else {                                 // Vh[hs][4096]  (M=n, N=hs)
    int t = wave - 5120;
    int hs0 = (t >> 8) << 4, n0 = (t & 255) << 4;
    v8f acc = gemm_k256(xt + (size_t)(n0 + col) * 256, Wvh + (size_t)(hs0 + col) * 256, hi);
    float b = bv[hs0 + col];
    v8h ph;
#pragma unroll
    for (int r = 0; r < 8; ++r) ph[r] = (_Float16)(acc[r] + b);
    *(v8h*)(Vh + (size_t)(hs0 + col) * N_SP + n0 + 8 * hi) = ph;
  }
}

// ---- Kernel 2: fused attention. WG = (head o, 16 rows). 16 waves, 256KB LDS. ----
__global__ __launch_bounds__(512) void k_attn(const _Float16* __restrict__ Qf,
                                              const _Float16* __restrict__ Kt,
                                              const _Float16* __restrict__ Vh,
                                              float* __restrict__ attn,
                                              _Float16* __restrict__ Obuf) {
  extern __shared__ float smem[];
  float* S = smem;                  // [16][PR]  logits (log2 domain)
  float* stats = smem + 16 * PR;    // [0..15]=rowmax, [16..31]=1/sum
  int w = threadIdx.x >> 5, lane = threadIdx.x & 31;
  int col = lane & 15, hi = lane >> 4;
  int n0 = blockIdx.x << 4, o = blockIdx.y;
  int mbase = w << 8;               // each wave owns 256 key columns

  const _Float16* arow = Qf + (size_t)o * (N_SP * 64) + (size_t)(n0 + col) * 64;
  v16h a0 = mk_afrag(arow, hi), a1 = mk_afrag(arow + 32, hi);

  // Phase 1: S = Qf . Kt  (K=64 -> 2 WMMAs per 16x16 tile), store f32 to LDS
#pragma unroll 4
  for (int mt = 0; mt < 16; ++mt) {
    int m0 = mbase + (mt << 4);
    const _Float16* bb = Kt + (size_t)(m0 + col) * 64;
    v16h b0 = *(const v16h*)(bb + hi * 16);
    v16h b1 = *(const v16h*)(bb + 32 + hi * 16);
    v8f acc = {};
    acc = wmma16(a0, b0, acc);
    acc = wmma16(a1, b1, acc);
#pragma unroll
    for (int r = 0; r < 8; ++r) S[(r + 8 * hi) * PR + m0 + col] = acc[r];
  }
  __syncthreads();

  // Phase 2: wave w computes max & sum(exp2) over its row
  {
    float* Srow = S + w * PR;
    float mv = -3.0e38f;
#pragma unroll 8
    for (int i = 0; i < 32; ++i) {
      v4f t = *(const v4f*)(Srow + (lane << 2) + (i << 7));
      mv = fmaxf(mv, fmaxf(fmaxf(t[0], t[1]), fmaxf(t[2], t[3])));
    }
    for (int d = 16; d; d >>= 1) mv = fmaxf(mv, __shfl_xor(mv, d, 32));
    float sum = 0.f;
#pragma unroll 8
    for (int i = 0; i < 32; ++i) {
      v4f t = *(const v4f*)(Srow + (lane << 2) + (i << 7));
      sum += exp2f(t[0] - mv) + exp2f(t[1] - mv) + exp2f(t[2] - mv) + exp2f(t[3] - mv);
    }
    for (int d = 16; d; d >>= 1) sum += __shfl_xor(sum, d, 32);
    if (lane == 0) { stats[w] = mv; stats[16 + w] = 1.0f / sum; }
  }
  __syncthreads();

  // Phase 3: normalize, stream probs to d_out (non-temporal: 268MB > 192MB L2),
  // and accumulate P.V via WMMA. S is read directly in contiguous A-fragment layout (k = 16*hi + s).
  int rowa = col;
  float rmax = stats[rowa], rinv = stats[16 + rowa];
  float* attnRow = attn + (size_t)(o * N_SP + n0 + rowa) * N_SP;
  v8f oacc = {};
#pragma unroll 2
  for (int blk = 0; blk < 8; ++blk) {
    int m0 = mbase + (blk << 5);
    const float* sp = S + rowa * PR + m0 + hi * 16;
    v4f s0 = *(const v4f*)(sp);
    v4f s1 = *(const v4f*)(sp + 4);
    v4f s2 = *(const v4f*)(sp + 8);
    v4f s3 = *(const v4f*)(sp + 12);
    v4f q0, q1, q2, q3;
#pragma unroll
    for (int j = 0; j < 4; ++j) {
      q0[j] = exp2f(s0[j] - rmax) * rinv;
      q1[j] = exp2f(s1[j] - rmax) * rinv;
      q2[j] = exp2f(s2[j] - rmax) * rinv;
      q3[j] = exp2f(s3[j] - rmax) * rinv;
    }
    float* gp = attnRow + m0 + hi * 16;
    __builtin_nontemporal_store(q0, (v4f*)(gp));
    __builtin_nontemporal_store(q1, (v4f*)(gp + 4));
    __builtin_nontemporal_store(q2, (v4f*)(gp + 8));
    __builtin_nontemporal_store(q3, (v4f*)(gp + 12));
    v16h pa;
#pragma unroll
    for (int j = 0; j < 4; ++j) {
      pa[j] = (_Float16)q0[j]; pa[4 + j] = (_Float16)q1[j];
      pa[8 + j] = (_Float16)q2[j]; pa[12 + j] = (_Float16)q3[j];
    }
    v16h bv16 = *(const v16h*)(Vh + (size_t)(o * 16 + col) * N_SP + m0 + hi * 16);
    oacc = wmma16(pa, bv16, oacc);
  }
  __syncthreads();

  // Phase 4: cross-wave reduce the 16x16 output tile (reuse S area), emit O f16
#pragma unroll
  for (int r = 0; r < 8; ++r) S[w * 256 + r * 32 + lane] = oacc[r];
  __syncthreads();
  if (threadIdx.x < 256) {
    int t = threadIdx.x;
    float s = 0.f;
#pragma unroll
    for (int ww = 0; ww < 16; ++ww) s += S[ww * 256 + t];
    int r = t >> 5, l = t & 31;
    int row = r + 8 * (l >> 4), cc = l & 15;
    Obuf[(size_t)(n0 + row) * 64 + o * 16 + cc] = (_Float16)s;
  }
}

// ---- Kernel 3: out = BN(Wo @ O^T). M=c, N=n -> coalesced f32 stores. ----
__global__ __launch_bounds__(256) void k_outproj(const _Float16* __restrict__ Obuf,
                                                 const _Float16* __restrict__ Woh,
                                                 const float* __restrict__ scaleBN,
                                                 const float* __restrict__ shiftBN,
                                                 float* __restrict__ out) {
  int wave = blockIdx.x * 8 + (threadIdx.x >> 5);
  int lane = threadIdx.x & 31, col = lane & 15, hi = lane >> 4;
  int ct = wave & 15, nt = wave >> 4;
  int c0 = ct << 4, n0 = nt << 4;
  const _Float16* Arow = Woh + (size_t)(c0 + col) * 64;
  const _Float16* Bcol = Obuf + (size_t)(n0 + col) * 64;
  v16h a0 = mk_afrag(Arow, hi), a1 = mk_afrag(Arow + 32, hi);
  v16h b0 = *(const v16h*)(Bcol + hi * 16);
  v16h b1 = *(const v16h*)(Bcol + 32 + hi * 16);
  v8f acc = {};
  acc = wmma16(a0, b0, acc);
  acc = wmma16(a1, b1, acc);
#pragma unroll
  for (int r = 0; r < 8; ++r) {
    int c = c0 + r + 8 * hi;
    __builtin_nontemporal_store(acc[r] * scaleBN[c] + shiftBN[c],
                                out + (size_t)c * N_SP + n0 + col);
  }
}

extern "C" void kernel_launch(void* const* d_in, const int* in_sizes, int n_in,
                              void* d_out, int out_size, void* d_ws, size_t ws_size,
                              hipStream_t stream) {
  const float* x     = (const float*)d_in[0];
  const float* Wq    = (const float*)d_in[1];
  const float* bq    = (const float*)d_in[2];
  const float* Wk    = (const float*)d_in[3];
  const float* bk    = (const float*)d_in[4];
  const float* Wv    = (const float*)d_in[5];
  const float* bv    = (const float*)d_in[6];
  const float* Wf    = (const float*)d_in[7];
  const float* Wo    = (const float*)d_in[8];
  const float* bo    = (const float*)d_in[9];
  const float* gamma = (const float*)d_in[10];
  const float* beta  = (const float*)d_in[11];
  const float* rmean = (const float*)d_in[12];
  const float* rvar  = (const float*)d_in[13];

  float* outp = (float*)d_out;                 // [256][64][64] f32
  float* attn = outp + (size_t)256 * 64 * 64;  // [4][4096][4096] f32

  unsigned char* w = (unsigned char*)d_ws;
  _Float16* xt    = (_Float16*)w; w += (size_t)N_SP * C_IN * 2;
  _Float16* Wq_s  = (_Float16*)w; w += (size_t)4 * 64 * 256 * 2;
  float*    bq_s  = (float*)w;    w += 256 * 4;
  _Float16* Wkh   = (_Float16*)w; w += 64 * 256 * 2;
  _Float16* Wvh   = (_Float16*)w; w += 64 * 256 * 2;
  _Float16* Woh   = (_Float16*)w; w += 256 * 64 * 2;
  float*    sBN   = (float*)w;    w += 256 * 4;
  float*    tBN   = (float*)w;    w += 256 * 4;
  _Float16* Qf    = (_Float16*)w; w += (size_t)4 * N_SP * 64 * 2;
  _Float16* Kt    = (_Float16*)w; w += (size_t)N_SP * 64 * 2;
  _Float16* Vh    = (_Float16*)w; w += (size_t)64 * N_SP * 2;
  _Float16* Obuf  = (_Float16*)w; w += (size_t)N_SP * 64 * 2;

  k_transpose_x<<<dim3(128, 8), 256, 0, stream>>>(x, xt);
  k_prep_w<<<64, 256, 0, stream>>>(Wq, bq, Wk, Wv, Wf, Wo, bo, gamma, beta, rmean, rvar,
                                   Wq_s, bq_s, Wkh, Wvh, Woh, sBN, tBN);
  k_qkv<<<768, 256, 0, stream>>>(xt, Wq_s, bq_s, Wkh, bk, Wvh, bv, Qf, Kt, Vh);

  const int attn_lds = (16 * PR + 32) * 4;     // 262,528 B of the 320 KB WGP LDS
  (void)hipFuncSetAttribute((const void*)k_attn,
                            hipFuncAttributeMaxDynamicSharedMemorySize, attn_lds);
  k_attn<<<dim3(256, 4), 512, attn_lds, stream>>>(Qf, Kt, Vh, attn, Obuf);

  k_outproj<<<512, 256, 0, stream>>>(Obuf, Woh, sBN, tBN, outp);
}
